// ModernAsymmetricHopfieldNetwork_17918603559170
// MI455X (gfx1250) — matrix operand order repaired and checked by reading, not compiled, not run on hardware
//
#include <hip/hip_runtime.h>
#include <hip/hip_bf16.h>

// ---------------------------------------------------------------------------
// Flash-attention formulation of the Modern Asymmetric Hopfield Network:
//   Q = s (8191x1024), K = X[:-1], V = X[1:]
//   out = softmax(Q K^T) V          (f32 in / f32 out, bf16 WMMA math)
//
// Block = 512 threads (16 waves), 32 q rows, 256 keys per iteration.
// Each wave owns 64 output columns -> acc = 64 VGPRs.  B operands are
// software-pipelined depth-8, and the schedule is pinned with
// sched_group_barrier so the prefetch distance survives the scheduler
// (otherwise loads sink to their consumers -> s_wait_loadcnt 0 stalls).
// ---------------------------------------------------------------------------

typedef __attribute__((ext_vector_type(16))) __bf16 v16bf;
typedef __attribute__((ext_vector_type(8)))  __bf16 v8bf;
typedef __attribute__((ext_vector_type(4)))  __bf16 v4bf;
typedef __attribute__((ext_vector_type(8)))  float  v8f;

#define DHEAD   1024          // feature dim N
#define QROWS   8191          // number of query rows (= P-1)
#define KEYS    8191          // number of keys       (= P-1)
#define KPAD    8192          // padded key count for V^T
#define BQ      32            // q rows per workgroup (2 sub-tiles of 16)
#define BK      256           // keys per iteration (16 waves x 16 keys)
#define NW      16            // waves per workgroup
#define NTHR    512           // threads per workgroup

#if defined(__has_builtin) && __has_builtin(__builtin_amdgcn_sched_group_barrier)
#define SGB(mask, n) __builtin_amdgcn_sched_group_barrier((mask), (n), 0)
#else
#define SGB(mask, n)
#endif
// masks: 0x008 = MFMA/WMMA, 0x020 = VMEM read, 0x100 = DS read

__device__ __forceinline__ v8f wmma_bf16(v16bf a, v16bf b, v8f c) {
    // D = A(16x32) * B(32x16) + C, f32 accumulate
    return __builtin_amdgcn_wmma_f32_16x16x32_bf16(
        /*neg_a=*/false, a, /*neg_b=*/false, b,
        /*c_mod=*/(short)0, c, /*reuse_a=*/false, /*reuse_b=*/false);
}

__device__ __forceinline__ v16bf cat8(v8bf lo, v8bf hi) {
    return __builtin_shufflevector(lo, hi,
        0,1,2,3,4,5,6,7,8,9,10,11,12,13,14,15);
}

// A-operand (16x32 bf16): lane l (M = l&15) holds K = {c..c+15}
// with c = 16*(l>>4), relative to the chunk base.  -> two contiguous 16B loads.
__device__ __forceinline__ v16bf load_A(const __bf16* rowbase, int half) {
    v8bf lo = *(const v8bf*)(rowbase + 16 * half);
    v8bf hi = *(const v8bf*)(rowbase + 16 * half + 8);
    return cat8(lo, hi);
}

// ---------------------------------------------------------------------------
// Prolog 1: f32 -> bf16 row-major convert (serves X->Kbf and s->Qbf)
// ---------------------------------------------------------------------------
__global__ __launch_bounds__(256) void cvt_bf16_kernel(
    const float* __restrict__ src, __bf16* __restrict__ dst, long n4)
{
    long i = (long)blockIdx.x * blockDim.x + threadIdx.x;
    if (i >= n4) return;
    float4 f = ((const float4*)src)[i];
    v4bf o;
    o[0] = (__bf16)f.x; o[1] = (__bf16)f.y;
    o[2] = (__bf16)f.z; o[3] = (__bf16)f.w;
    ((v4bf*)dst)[i] = o;
}

// ---------------------------------------------------------------------------
// Prolog 2: V^T build.  Vt[n][k] = bf16(X[k+1][n]), k in [0,KPAD), zero-pad
// the k==KEYS column.  Tiled 32x32 transpose through LDS, coalesced both ways.
// ---------------------------------------------------------------------------
__global__ __launch_bounds__(256) void transpose_v_kernel(
    const float* __restrict__ X, __bf16* __restrict__ Vt)
{
    __shared__ float tile[32][33];
    const int kb = blockIdx.x * 32;       // key tile
    const int nb = blockIdx.y * 32;       // feature tile
    const int tx = threadIdx.x;           // 0..31
    const int ty = threadIdx.y;           // 0..7
#pragma unroll
    for (int j = 0; j < 32; j += 8) {
        int k = kb + ty + j;
        int n = nb + tx;
        float v = (k < KEYS) ? X[(size_t)(k + 1) * DHEAD + n] : 0.0f;
        tile[ty + j][tx] = v;
    }
    __syncthreads();
#pragma unroll
    for (int j = 0; j < 32; j += 8) {
        int n = nb + ty + j;
        int k = kb + tx;
        Vt[(size_t)n * KPAD + k] = (__bf16)tile[tx][ty + j];
    }
}

// ---------------------------------------------------------------------------
// Main fused flash-attention kernel.
// ---------------------------------------------------------------------------
__global__ __launch_bounds__(NTHR) void mahn_flash_kernel(
    const __bf16* __restrict__ Kbf,   // 8192 x 1024  (X in bf16; rows 0..8190 are K)
    const __bf16* __restrict__ Qbf,   // 8191 x 1024  (s in bf16)
    const __bf16* __restrict__ Vt,    // 1024 x 8192  (V^T, col 8191 zeroed)
    float*        __restrict__ out)   // 8191 x 1024
{
    extern __shared__ char smem[];
    __bf16* Qs     = (__bf16*)(smem);                           // 32x1024 = 64 KB
    float*  Sraw   = (float*) (smem + 65536);                   // 32x256  = 32 KB
    __bf16* Pbf    = (__bf16*)(smem + 65536 + 32768);           // 32x256  = 16 KB
    float*  red    = (float*) (smem + 65536 + 32768 + 16384);   // 32x16   =  2 KB
    float*  mI     = (float*) (smem + 65536 + 32768 + 16384 + 2048);  // 32
    float*  lI     = mI + 32;                                          // 32
    float*  alphaI = lI + 32;                                          // 32

    const int tid   = threadIdx.x;
    const int wave  = tid >> 5;       // 0..15
    const int lane  = tid & 31;
    const int half  = lane >> 4;      // 0 or 1
    const int l15   = lane & 15;
    const int q0    = blockIdx.x * BQ;

    // ---- load Q tile into LDS as bf16 (rows clamped at the ragged edge) ----
    for (int idx = tid; idx < BQ * (DHEAD / 8); idx += NTHR) {
        int row  = idx >> 7;                // / (DHEAD/8)
        int off  = (idx & 127) * 8;
        int qrow = q0 + row; if (qrow > QROWS - 1) qrow = QROWS - 1;
        *(uint4*)&Qs[row * DHEAD + off] =
            *(const uint4*)&Qbf[(size_t)qrow * DHEAD + off];
    }
    if (tid < BQ) { mI[tid] = -INFINITY; lI[tid] = 0.0f; }

    v8f acc[2][4];                    // 2 q-subtiles x 4 col tiles = 64 VGPRs
#pragma unroll
    for (int u = 0; u < 2; ++u)
#pragma unroll
        for (int t = 0; t < 4; ++t) acc[u][t] = (v8f)0.0f;

    __syncthreads();

    for (int k0 = 0; k0 < KEYS; k0 += BK) {
        // ================= phase 1: S(32 x 16) for this wave's keys =======
        // B operand: lane l = key (k0 + 16*wave + l15), contiguous 16 halves
        // at column n0 + 16*half of that key's row.
        const int keyrow = k0 + wave * 16 + l15;            // <= 8191, in-bounds
        const __bf16* kptr  = Kbf + (size_t)keyrow * DHEAD + half * 16;
        const __bf16* qrow0 = Qs  + (size_t)l15 * DHEAD;            // sub-tile 0
        const __bf16* qrow1 = Qs  + (size_t)(16 + l15) * DHEAD;     // sub-tile 1

        v8f s0 = (v8f)0.0f, s1 = (v8f)0.0f;
        {
            v16bf bq[8];                       // depth-8 rotating B buffer
#pragma unroll
            for (int i = 0; i < 8; ++i)
                bq[i] = *(const v16bf*)(kptr + i * 32);
#pragma unroll
            for (int c = 0; c < 32; ++c) {
                const int n0 = c * 32;
                v16bf b = bq[c & 7];
                if (c + 8 < 32)
                    bq[c & 7] = *(const v16bf*)(kptr + (c + 8) * 32);
                v16bf a0 = load_A(qrow0 + n0, half);
                v16bf a1 = load_A(qrow1 + n0, half);
                s0 = wmma_bf16(a0, b, s0);
                s1 = wmma_bf16(a1, b, s1);
                // pin pipeline: prefetch loads interleave with this step's WMMAs
                if (c + 8 < 32) SGB(0x020, 2);   // 2x global_load_b128
                SGB(0x100, 4);                   // 4x ds_load_b128 (A operands)
                SGB(0x008, 2);                   // 2x wmma
            }
        }
        // scatter S tiles to LDS: row = (u*16) + r + 8*half, col = 16*wave+l15
        {
            const int col = wave * 16 + l15;
#pragma unroll
            for (int r = 0; r < 8; ++r) {
                Sraw[(r + 8 * half) * BK + col]      = s0[r];
                Sraw[(16 + r + 8 * half) * BK + col] = s1[r];
            }
        }
        __syncthreads();   // (A)

        // ================= online softmax over the 32x256 block ============
        const int srow = tid >> 4;          // 0..31
        const int seg  = tid & 15;          // 0..15, 16 cols each
        float vals[16];
        {
            float mloc = -INFINITY;
#pragma unroll
            for (int i = 0; i < 16; ++i) {
                int col = seg * 16 + i;
                float v = Sraw[srow * BK + col];
                if (k0 + col >= KEYS) v = -INFINITY;     // mask ragged keys
                vals[i] = v;
                mloc = fmaxf(mloc, v);
            }
            red[srow * 16 + seg] = mloc;
        }
        __syncthreads();   // (B)
        if (tid < BQ) {
            float m = -INFINITY;
#pragma unroll
            for (int i = 0; i < 16; ++i) m = fmaxf(m, red[tid * 16 + i]);
            float mprev = mI[tid];
            float mnew  = fmaxf(mprev, m);
            alphaI[tid] = __expf(mprev - mnew);   // 0 on first pass
            mI[tid]     = mnew;
        }
        __syncthreads();   // (C)
        {
            float mnew = mI[srow];
            float ssum = 0.0f;
#pragma unroll
            for (int i = 0; i < 16; ++i) {
                int col = seg * 16 + i;
                float p = (vals[i] == -INFINITY) ? 0.0f : __expf(vals[i] - mnew);
                ssum += p;
                Pbf[srow * BK + col] = (__bf16)p;
            }
            red[srow * 16 + seg] = ssum;
        }
        __syncthreads();   // (D)
        if (tid < BQ) {
            float s = 0.0f;
#pragma unroll
            for (int i = 0; i < 16; ++i) s += red[tid * 16 + i];
            lI[tid] = lI[tid] * alphaI[tid] + s;
        }

        // ================= phase 2: O = alpha*O + P V  (64 cols/wave) ======
#pragma unroll
        for (int u = 0; u < 2; ++u) {
            float al[8];
#pragma unroll
            for (int r = 0; r < 8; ++r) al[r] = alphaI[u * 16 + r + 8 * half];
#pragma unroll
            for (int t = 0; t < 4; ++t)
#pragma unroll
                for (int r = 0; r < 8; ++r) acc[u][t][r] *= al[r];
        }
        {
            // flattened (kc, t) loop: step s -> kc = s>>2 (x32 keys), t = s&3
            // B addr for (kc, t): vbase + t*16*KPAD + kc*32   (compile-time offs)
            const __bf16* vbase = Vt + (size_t)(wave * 64 + l15) * KPAD
                                     + k0 + half * 16;
            v16bf bq[8];                       // depth-8 rotating B buffer
#pragma unroll
            for (int i = 0; i < 8; ++i)
                bq[i] = *(const v16bf*)(vbase
                    + (size_t)(i & 3) * 16 * KPAD + (i >> 2) * 32);
            v16bf a0, a1;
#pragma unroll
            for (int s = 0; s < 32; ++s) {
                const int kc = (s >> 2) * 32;
                const int t  = s & 3;
                if (t == 0) {
                    a0 = load_A(&Pbf[l15 * BK + kc], half);
                    a1 = load_A(&Pbf[(16 + l15) * BK + kc], half);
                }
                v16bf b = bq[s & 7];
                if (s + 8 < 32) {
                    const int s2 = s + 8;
                    bq[s & 7] = *(const v16bf*)(vbase
                        + (size_t)(s2 & 3) * 16 * KPAD + (s2 >> 2) * 32);
                }
                acc[0][t] = wmma_bf16(a0, b, acc[0][t]);
                acc[1][t] = wmma_bf16(a1, b, acc[1][t]);
                // pin pipeline: keep the depth-8 prefetch ahead of consumers
                if (s + 8 < 32) SGB(0x020, 2);   // 2x global_load_b128
                if (t == 0)     SGB(0x100, 4);   // 4x ds_load_b128 (A operands)
                SGB(0x008, 2);                   // 2x wmma
            }
        }
        __syncthreads();   // (E) protect LDS reuse
    }

    // ================= epilogue: out = O / l ==============================
#pragma unroll
    for (int u = 0; u < 2; ++u) {
        float linv[8];
#pragma unroll
        for (int r = 0; r < 8; ++r) linv[r] = 1.0f / lI[u * 16 + r + 8 * half];
#pragma unroll
        for (int t = 0; t < 4; ++t) {
            int col = wave * 64 + t * 16 + l15;
#pragma unroll
            for (int r = 0; r < 8; ++r) {
                int q = q0 + u * 16 + r + 8 * half;
                if (q < QROWS)
                    out[(size_t)q * DHEAD + col] = acc[u][t][r] * linv[r];
            }
        }
    }
}

// ---------------------------------------------------------------------------
extern "C" void kernel_launch(void* const* d_in, const int* in_sizes, int n_in,
                              void* d_out, int out_size, void* d_ws, size_t ws_size,
                              hipStream_t stream) {
    const float* X = (const float*)d_in[0];   // 8192 x 1024 f32
    const float* s = (const float*)d_in[1];   // 8191 x 1024 f32
    float* out = (float*)d_out;               // 8191 x 1024 f32

    // workspace layout (bf16): Xbf | sbf | Vt   (~48 MB total)
    __bf16* Xbf = (__bf16*)d_ws;
    __bf16* Qbf = Xbf + (size_t)8192 * DHEAD;
    __bf16* Vt  = Qbf + (size_t)QROWS * DHEAD;

    // Prolog: converts + V^T build (memory-bound, ~5 us at 23 TB/s)
    {
        long n4x = (long)8192 * DHEAD / 4;
        long n4s = (long)QROWS * DHEAD / 4;
        cvt_bf16_kernel<<<(int)((n4x + 255) / 256), 256, 0, stream>>>(X, Xbf, n4x);
        cvt_bf16_kernel<<<(int)((n4s + 255) / 256), 256, 0, stream>>>(s, Qbf, n4s);
        dim3 tg(KPAD / 32, DHEAD / 32);
        transpose_v_kernel<<<tg, dim3(32, 8), 0, stream>>>(X, Vt);
    }

    // Fused flash attention: 256 q-blocks x 512 threads, ~117 KB dynamic LDS
    {
        const int nblocks = (QROWS + BQ - 1) / BQ;             // 256
        const size_t lds = 65536 + 32768 + 16384 + 2048 + 3 * 32 * sizeof(float);
        mahn_flash_kernel<<<nblocks, NTHR, lds, stream>>>(Xbf, Qbf, Vt, out);
    }
}
